// MixerLayers_3874060501268
// MI455X (gfx1250) — compile-verified, matrix operand order checked by reading, not run on hardware
//
#include <hip/hip_runtime.h>
#include <hip/hip_bf16.h>
#include <math.h>

// ---- problem constants (match reference) ----
constexpr int CB   = 2;          // batch
constexpr int CL   = 1024;       // seq len
constexpr int CD   = 768;        // model dim
constexpr int CDI  = 1536;       // inner dim
constexpr int C2DI = 3072;       // 2*inner
constexpr int CDS  = 16;         // state dim
constexpr int CDC  = 4;          // conv width
constexpr int CR   = 48;         // dt rank
constexpr int CXN  = 80;         // R + 2*DS
constexpr int CM   = CB * CL;    // 2048 tokens
constexpr int CDEPTH = 4;
constexpr float CEPS = 1e-5f;

typedef _Float16 half8 __attribute__((ext_vector_type(8)));
typedef _Float16 v16h  __attribute__((ext_vector_type(16)));
typedef float    v8f   __attribute__((ext_vector_type(8)));

// Load a 16x32 f16 fragment (rows x K) per CDNA5 WMMA A/B layout:
//   lane<16 : row=lane,    K in {k0..k0+7, k0+16..k0+23}
//   lane>=16: row=lane-16, K in {k0+8..k0+15, k0+24..k0+31}
// -> two contiguous 16B chunks per lane.
__device__ __forceinline__ v16h load_frag16x32(const _Float16* __restrict__ base,
                                               int row0, int k0, int ldk, int lane) {
  const int r  = lane & 15;
  const int hi = lane >> 4;
  const _Float16* p = base + (size_t)(row0 + r) * ldk + (k0 + hi * 8);
  half8 c0 = *(const half8*)(p);
  half8 c1 = *(const half8*)(p + 16);
  return __builtin_shufflevector(c0, c1, 0,1,2,3,4,5,6,7,8,9,10,11,12,13,14,15);
}

// C[m,n] = sum_k A[m,k] * W[n,k]   (A: MxK f16 row-major, W: NxK f16 row-major)
// One wave computes a 16 x (16*NT) strip. K-loop unrolled x2 as a register
// ping-pong (compute buf0 while loading buf1 and vice versa) -> no v_mov
// rotation, loads overlap wmmas. Requires K % 64 == 0 (768/1536 both are).
template <int NT>
__global__ void __launch_bounds__(32)
gemm_wmma_f16(const _Float16* __restrict__ A, const _Float16* __restrict__ W,
              float* __restrict__ C, int M, int N, int K) {
  const int m0   = blockIdx.x * 16;
  const int n0   = blockIdx.y * 16 * NT;
  const int lane = threadIdx.x & 31;

  v8f acc[NT];
#pragma unroll
  for (int i = 0; i < NT; ++i) acc[i] = (v8f){};

  // prologue: buffer 0 holds k = 0
  v16h a0 = load_frag16x32(A, m0, 0, K, lane);
  v16h b0[NT];
#pragma unroll
  for (int i = 0; i < NT; ++i) b0[i] = load_frag16x32(W, n0 + 16 * i, 0, K, lane);

  for (int k0 = 0; k0 < K; k0 += 64) {
    // stage 1: issue loads for k0+32 into buffer 1, compute on buffer 0
    __builtin_prefetch(A + (size_t)(m0 + (lane & 15)) * K + k0 + 128, 0, 3);
    v16h a1 = load_frag16x32(A, m0, k0 + 32, K, lane);
    v16h b1[NT];
#pragma unroll
    for (int i = 0; i < NT; ++i)
      b1[i] = load_frag16x32(W, n0 + 16 * i, k0 + 32, K, lane);
#pragma unroll
    for (int i = 0; i < NT; ++i)
      acc[i] = __builtin_amdgcn_wmma_f32_16x16x32_f16(false, a0, false, b0[i],
                                                      (short)0, acc[i], false, false);
    // stage 2: issue loads for k0+64 into buffer 0, compute on buffer 1
    if (k0 + 64 < K) {
      a0 = load_frag16x32(A, m0, k0 + 64, K, lane);
#pragma unroll
      for (int i = 0; i < NT; ++i)
        b0[i] = load_frag16x32(W, n0 + 16 * i, k0 + 64, K, lane);
    }
#pragma unroll
    for (int i = 0; i < NT; ++i)
      acc[i] = __builtin_amdgcn_wmma_f32_16x16x32_f16(false, a1, false, b1[i],
                                                      (short)0, acc[i], false, false);
  }

  // C/D layout: VGPR v -> M = v + 8*(lane>=16); N = lane&15
  const int n  = lane & 15;
  const int mb = (lane >> 4) * 8;
#pragma unroll
  for (int i = 0; i < NT; ++i)
#pragma unroll
    for (int v = 0; v < 8; ++v)
      C[(size_t)(m0 + mb + v) * N + (n0 + 16 * i + n)] = acc[i][v];
}

__global__ void f32_to_f16_kernel(const float* __restrict__ src,
                                  _Float16* __restrict__ dst, int n) {
  int i = blockIdx.x * blockDim.x + threadIdx.x;
  if (i < n) dst[i] = (_Float16)src[i];
}

// residual accumulate + LayerNorm -> f16 activations for WMMA
__global__ void __launch_bounds__(256)
add_ln_kernel(const float* __restrict__ h, float* __restrict__ resid,
              const float* __restrict__ w, const float* __restrict__ b,
              _Float16* __restrict__ hn16, int first) {
  const int row = blockIdx.x;
  const int tid = threadIdx.x;
  __shared__ float s1[256], s2[256];
  float v[CD / 256];
  float lsum = 0.f, lsq = 0.f;
#pragma unroll
  for (int j = 0; j < CD / 256; ++j) {
    int c = tid + j * 256;
    float x = h[(size_t)row * CD + c];
    if (!first) x += resid[(size_t)row * CD + c];
    resid[(size_t)row * CD + c] = x;
    v[j] = x; lsum += x; lsq += x * x;
  }
  s1[tid] = lsum; s2[tid] = lsq;
  __syncthreads();
  for (int off = 128; off > 0; off >>= 1) {
    if (tid < off) { s1[tid] += s1[tid + off]; s2[tid] += s2[tid + off]; }
    __syncthreads();
  }
  const float mu   = s1[0] / (float)CD;
  const float var  = s2[0] / (float)CD - mu * mu;
  const float rstd = rsqrtf(var + CEPS);
#pragma unroll
  for (int j = 0; j < CD / 256; ++j) {
    int c = tid + j * 256;
    hn16[(size_t)row * CD + c] = (_Float16)(((v[j] - mu) * rstd) * w[c] + b[c]);
  }
}

// final: out = LN(h + resid) in f32
__global__ void __launch_bounds__(256)
final_ln_kernel(const float* __restrict__ h, const float* __restrict__ resid,
                const float* __restrict__ w, const float* __restrict__ b,
                float* __restrict__ out) {
  const int row = blockIdx.x;
  const int tid = threadIdx.x;
  __shared__ float s1[256], s2[256];
  float v[CD / 256];
  float lsum = 0.f, lsq = 0.f;
#pragma unroll
  for (int j = 0; j < CD / 256; ++j) {
    int c = tid + j * 256;
    float x = h[(size_t)row * CD + c] + resid[(size_t)row * CD + c];
    v[j] = x; lsum += x; lsq += x * x;
  }
  s1[tid] = lsum; s2[tid] = lsq;
  __syncthreads();
  for (int off = 128; off > 0; off >>= 1) {
    if (tid < off) { s1[tid] += s1[tid + off]; s2[tid] += s2[tid + off]; }
    __syncthreads();
  }
  const float mu   = s1[0] / (float)CD;
  const float var  = s2[0] / (float)CD - mu * mu;
  const float rstd = rsqrtf(var + CEPS);
#pragma unroll
  for (int j = 0; j < CD / 256; ++j) {
    int c = tid + j * 256;
    out[(size_t)row * CD + c] = ((v[j] - mu) * rstd) * w[c] + b[c];
  }
}

// causal depthwise conv (width 4) + SiLU; reads x-half of xz, writes f32 + f16
__global__ void __launch_bounds__(256)
conv_silu_kernel(const float* __restrict__ xz,
                 const float* __restrict__ cw, const float* __restrict__ cb,
                 float* __restrict__ xm, _Float16* __restrict__ xm16) {
  int idx = blockIdx.x * blockDim.x + threadIdx.x;
  if (idx >= CM * CDI) return;
  int row = idx / CDI, c = idx % CDI;
  int bb = row / CL, t = row % CL;
  float acc = cb[c];
#pragma unroll
  for (int j = 0; j < CDC; ++j) {
    int tt = t - (CDC - 1) + j;
    if (tt >= 0)
      acc += xz[(size_t)(bb * CL + tt) * C2DI + c] * cw[c * CDC + j];
  }
  float s = acc / (1.f + expf(-acc));   // silu
  xm[idx]   = s;
  xm16[idx] = (_Float16)s;
}

// dt = softplus(dbc[:, :48] @ dtw^T + dtb)   (tiny K=48 -> VALU)
__global__ void __launch_bounds__(256)
dt_softplus_kernel(const float* __restrict__ dbc,
                   const float* __restrict__ dtw, const float* __restrict__ dtb,
                   float* __restrict__ dt) {
  int idx = blockIdx.x * blockDim.x + threadIdx.x;
  if (idx >= CM * CDI) return;
  int row = idx / CDI, d = idx % CDI;
  const float* dr = dbc + (size_t)row * CXN;
  const float* wr = dtw + (size_t)d * CR;
  float s = dtb[d];
#pragma unroll 8
  for (int r = 0; r < CR; ++r) s += dr[r] * wr[r];
  dt[idx] = (s > 20.f) ? s : log1pf(expf(s));
}

// selective scan: one lane per (b, d, n) state element; 16-lane shuffle-reduce
// for y = sum_n h*C; fuses +D*x and SiLU(z) gate; writes y as f16 for out_proj.
__global__ void __launch_bounds__(256)
scan_kernel(const float* __restrict__ dbc, const float* __restrict__ dtv,
            const float* __restrict__ xm, const float* __restrict__ xz,
            const float* __restrict__ A_log, const float* __restrict__ Dp,
            _Float16* __restrict__ y16) {
  int gid = blockIdx.x * blockDim.x + threadIdx.x;   // 0 .. CB*CDI*CDS
  int n  = gid & (CDS - 1);
  int d  = (gid / CDS) % CDI;
  int bb = gid / (CDS * CDI);
  const float A  = -expf(A_log[(size_t)d * CDS + n]);
  const float Dv = Dp[d];
  float h = 0.f;
  for (int t = 0; t < CL; ++t) {
    const size_t row = (size_t)bb * CL + t;
    const float dt = dtv[row * CDI + d];        // broadcast within 16-lane group
    const float x  = xm[row * CDI + d];
    const float Bt = dbc[row * CXN + CR + n];         // coalesced across group
    const float Ct = dbc[row * CXN + CR + CDS + n];
    h = expf(dt * A) * h + (dt * x) * Bt;
    float y = h * Ct;
    y += __shfl_xor(y, 1, 32);
    y += __shfl_xor(y, 2, 32);
    y += __shfl_xor(y, 4, 32);
    y += __shfl_xor(y, 8, 32);
    if (n == 0) {
      float z  = xz[row * C2DI + CDI + d];
      float yy = (y + Dv * x) * (z / (1.f + expf(-z)));
      y16[row * CDI + d] = (_Float16)yy;
    }
  }
}

extern "C" void kernel_launch(void* const* d_in, const int* in_sizes, int n_in,
                              void* d_out, int out_size, void* d_ws, size_t ws_size,
                              hipStream_t stream) {
  const float* hidden = (const float*)d_in[0];
  const float* in_w   = (const float*)d_in[1];
  const float* conv_w = (const float*)d_in[2];
  const float* conv_b = (const float*)d_in[3];
  const float* xp_w   = (const float*)d_in[4];
  const float* dt_w   = (const float*)d_in[5];
  const float* dt_b   = (const float*)d_in[6];
  const float* A_log  = (const float*)d_in[7];
  const float* Dpar   = (const float*)d_in[8];
  const float* out_w  = (const float*)d_in[9];
  const float* ln_w   = (const float*)d_in[10];
  const float* ln_b   = (const float*)d_in[11];
  const float* nf_w   = (const float*)d_in[12];
  const float* nf_b   = (const float*)d_in[13];
  float* out = (float*)d_out;

  char* ws = (char*)d_ws;
  size_t off = 0;
  auto alloc = [&](size_t bytes) -> void* {
    void* p = ws + off;
    off += (bytes + 255) & ~(size_t)255;
    return p;
  };

  float*    resid = (float*)alloc((size_t)CM * CD * 4);
  float*    hbuf  = (float*)alloc((size_t)CM * CD * 4);
  _Float16* hn16  = (_Float16*)alloc((size_t)CM * CD * 2);
  float*    xz    = (float*)alloc((size_t)CM * C2DI * 4);
  float*    xm    = (float*)alloc((size_t)CM * CDI * 4);
  _Float16* xm16  = (_Float16*)alloc((size_t)CM * CDI * 2);
  float*    dbc   = (float*)alloc((size_t)CM * CXN * 4);
  float*    dt    = (float*)alloc((size_t)CM * CDI * 4);
  _Float16* y16   = (_Float16*)alloc((size_t)CM * CDI * 2);
  _Float16* w_in16  = (_Float16*)alloc((size_t)CDEPTH * C2DI * CD * 2);
  _Float16* w_xp16  = (_Float16*)alloc((size_t)CDEPTH * CXN * CDI * 2);
  _Float16* w_out16 = (_Float16*)alloc((size_t)CDEPTH * CD * CDI * 2);

  // convert all layer weights to f16 once per call (deterministic)
  {
    int n1 = CDEPTH * C2DI * CD;
    f32_to_f16_kernel<<<(n1 + 255) / 256, 256, 0, stream>>>(in_w, w_in16, n1);
    int n2 = CDEPTH * CXN * CDI;
    f32_to_f16_kernel<<<(n2 + 255) / 256, 256, 0, stream>>>(xp_w, w_xp16, n2);
    int n3 = CDEPTH * CD * CDI;
    f32_to_f16_kernel<<<(n3 + 255) / 256, 256, 0, stream>>>(out_w, w_out16, n3);
  }

  for (int i = 0; i < CDEPTH; ++i) {
    const float* h_src = (i == 0) ? hidden : hbuf;

    add_ln_kernel<<<CM, 256, 0, stream>>>(h_src, resid,
                                          ln_w + (size_t)i * CD, ln_b + (size_t)i * CD,
                                          hn16, i == 0);

    // xz = hn @ in_w^T : (2048 x 3072 x 768), 16x64 strip per wave
    gemm_wmma_f16<4><<<dim3(CM / 16, C2DI / 64), 32, 0, stream>>>(
        hn16, w_in16 + (size_t)i * C2DI * CD, xz, CM, C2DI, CD);

    conv_silu_kernel<<<(CM * CDI + 255) / 256, 256, 0, stream>>>(
        xz, conv_w + (size_t)i * CDI * CDC, conv_b + (size_t)i * CDI, xm, xm16);

    // dbc = xm @ xp_w^T : (2048 x 80 x 1536), N=80 -> single-tile waves
    gemm_wmma_f16<1><<<dim3(CM / 16, CXN / 16), 32, 0, stream>>>(
        xm16, w_xp16 + (size_t)i * CXN * CDI, dbc, CM, CXN, CDI);

    dt_softplus_kernel<<<(CM * CDI + 255) / 256, 256, 0, stream>>>(
        dbc, dt_w + (size_t)i * CDI * CR, dt_b + (size_t)i * CDI, dt);

    scan_kernel<<<(CB * CDI * CDS) / 256, 256, 0, stream>>>(
        dbc, dt, xm, xz, A_log + (size_t)i * CDI * CDS, Dpar + (size_t)i * CDI, y16);

    // h = y @ out_w^T : (2048 x 768 x 1536), 16x64 strip per wave
    gemm_wmma_f16<4><<<dim3(CM / 16, CD / 64), 32, 0, stream>>>(
        y16, w_out16 + (size_t)i * CD * CDI, hbuf, CM, CD, CDI);
  }

  final_ln_kernel<<<CM, 256, 0, stream>>>(hbuf, resid, nf_w, nf_b, out);
}
